// QParamsLinear_90692529423048
// MI455X (gfx1250) — 8-bit matrix operand order repaired and checked by reading, compiled, not run on hardware
//
#include <hip/hip_runtime.h>
#include <hip/hip_bf16.h>

typedef __attribute__((ext_vector_type(16))) int          v16i;
typedef __attribute__((ext_vector_type(8)))  float        v8f;
typedef __attribute__((ext_vector_type(4)))  float        vf4;
typedef __attribute__((ext_vector_type(2)))  unsigned int vu2;
typedef __attribute__((ext_vector_type(4)))  unsigned int vu4;

#define AS_GLOBAL __attribute__((address_space(1)))

typedef const AS_GLOBAL unsigned char* gk_cu8;
typedef const AS_GLOBAL vu2*           gk_cu2;
typedef const AS_GLOBAL vu4*           gk_cu4;
typedef const AS_GLOBAL float*         gk_cf32;
typedef const AS_GLOBAL vf4*           gk_cf4;
typedef AS_GLOBAL float*               g_f32;
typedef AS_GLOBAL unsigned int*        g_u32;

#define FP8_MAX_F 448.0f

__device__ __forceinline__ unsigned int pack4_e4m3(float f0, float f1, float f2, float f3) {
#if __has_builtin(__builtin_amdgcn_cvt_pk_fp8_f32)
    int packed = 0;
    packed = __builtin_amdgcn_cvt_pk_fp8_f32(f0, f1, packed, false);
    packed = __builtin_amdgcn_cvt_pk_fp8_f32(f2, f3, packed, true);
    return (unsigned int)packed;
#else
    float v[4] = {f0, f1, f2, f3};
    unsigned int r = 0;
#pragma unroll
    for (int i = 0; i < 4; ++i) {
        float f = v[i];
        unsigned int u = __float_as_uint(f);
        unsigned int sgn = (u >> 31) << 7;
        float a = fabsf(f);
        unsigned int q;
        if (a < 0.015625f) {
            q = (unsigned int)(a * 512.0f + 0.5f);
        } else {
            unsigned int bits = __float_as_uint(a);
            unsigned int lsb = (bits >> 20) & 1u;
            bits += 0x7FFFFu + lsb;
            int e = (int)(bits >> 23) - 127;
            unsigned int m = (bits >> 20) & 0x7u;
            q = ((unsigned int)(e + 7) << 3) | m;
            if (q > 0x7Eu) q = 0x7Eu;
        }
        r |= (sgn | q) << (8 * i);
    }
    return r;
#endif
}

__global__ void quant_fp8_kernel(const float* __restrict__ in,
                                 unsigned int* __restrict__ out,
                                 const float* __restrict__ scale,
                                 long long n4) {
    long long i = (long long)blockIdx.x * blockDim.x + threadIdx.x;
    if (i >= n4) return;
    gk_cf4  ing  = (gk_cf4)(unsigned long long)in;
    g_u32   outg = (g_u32)(unsigned long long)out;
    gk_cf32 sg   = (gk_cf32)(unsigned long long)scale;
    const float inv = 1.0f / sg[0];
    vf4 v = __builtin_nontemporal_load(ing + i);
    float f0 = fminf(fmaxf(v.x * inv, -FP8_MAX_F), FP8_MAX_F);
    float f1 = fminf(fmaxf(v.y * inv, -FP8_MAX_F), FP8_MAX_F);
    float f2 = fminf(fmaxf(v.z * inv, -FP8_MAX_F), FP8_MAX_F);
    float f3 = fminf(fmaxf(v.w * inv, -FP8_MAX_F), FP8_MAX_F);
    outg[i] = pack4_e4m3(f0, f1, f2, f3);
}

__global__ __launch_bounds__(256, 1)
void fp8_wmma_gemm_kernel(const unsigned char* __restrict__ Aq,
                          const unsigned char* __restrict__ Bq,
                          const float* __restrict__ bias,
                          const float* __restrict__ s_in,
                          const float* __restrict__ s_w,
                          float* __restrict__ out,
                          int M, int N, int K) {
    const int lane = threadIdx.x & 31;
    const int wave = threadIdx.x >> 5;
    const int wm = wave >> 2;
    const int wn = wave & 3;
    const int rowBase = blockIdx.y * 128 + wm * 64;
    const int colBase = blockIdx.x * 256 + wn * 64;
    const int lmod = lane & 15;
    const int lhi  = lane >> 4;

    gk_cu8  Ag = (gk_cu8)(unsigned long long)Aq;
    gk_cu8  Bg = (gk_cu8)(unsigned long long)Bq;
    gk_cf32 biasg = (gk_cf32)(unsigned long long)bias;
    gk_cf32 sing  = (gk_cf32)(unsigned long long)s_in;
    gk_cf32 swg   = (gk_cf32)(unsigned long long)s_w;
    g_f32   og    = (g_f32)(unsigned long long)out;

    v8f acc[4][4];
#pragma unroll
    for (int i = 0; i < 4; ++i)
#pragma unroll
        for (int j = 0; j < 4; ++j) {
            v8f z = {0.f, 0.f, 0.f, 0.f, 0.f, 0.f, 0.f, 0.f};
            acc[i][j] = z;
        }

    gk_cu8 aptr[4];
    gk_cu8 bptr[4];
#pragma unroll
    for (int mt = 0; mt < 4; ++mt)
        aptr[mt] = Ag + (size_t)(rowBase + mt * 16 + lmod) * K + lhi * 8;
#pragma unroll
    for (int nt = 0; nt < 4; ++nt)
        bptr[nt] = Bg + (size_t)(colBase + nt * 16 + lmod) * K + lhi * 8;

    for (int k0 = 0; k0 < K; k0 += 128) {
        v16i afrag[4];
        v16i bfrag[4];
#pragma unroll
        for (int mt = 0; mt < 4; ++mt) {
#pragma unroll
            for (int p = 0; p < 8; ++p) {
                vu2 d = *(gk_cu2)(aptr[mt] + p * 16);
                afrag[mt][2 * p]     = (int)d.x;
                afrag[mt][2 * p + 1] = (int)d.y;
            }
            aptr[mt] += 128;
        }
#pragma unroll
        for (int nt = 0; nt < 4; ++nt) {
#pragma unroll
            for (int c = 0; c < 8; ++c) {
                vu2 d = *(gk_cu2)(bptr[nt] + c * 16);
                bfrag[nt][2 * c]     = (int)d.x;
                bfrag[nt][2 * c + 1] = (int)d.y;
            }
            bptr[nt] += 128;
        }
#pragma unroll
        for (int mt = 0; mt < 4; ++mt)
#pragma unroll
            for (int nt = 0; nt < 4; ++nt)
                acc[mt][nt] = __builtin_amdgcn_wmma_f32_16x16x128_fp8_fp8(
                    afrag[mt], bfrag[nt], (short)0, acc[mt][nt],
                     false,  false);
    }

    const float s = sing[0] * swg[0];
#pragma unroll
    for (int nt = 0; nt < 4; ++nt) {
        const int n = colBase + nt * 16 + lmod;
        const float bn = biasg[n];
#pragma unroll
        for (int mt = 0; mt < 4; ++mt) {
#pragma unroll
            for (int r = 0; r < 8; ++r) {
                const int m = rowBase + mt * 16 + lhi * 8 + r;
                __builtin_nontemporal_store(acc[mt][nt][r] * s + bn,
                                            og + (size_t)m * N + n);
            }
        }
    }
}

extern "C" void kernel_launch(void* const* d_in, const int* in_sizes, int n_in,
                              void* d_out, int out_size, void* d_ws, size_t ws_size,
                              hipStream_t stream) {
    const float* x      = (const float*)d_in[0];
    const float* weight = (const float*)d_in[1];
    const float* bias   = (const float*)d_in[2];
    const float* s_in   = (const float*)d_in[3];
    const float* s_w    = (const float*)d_in[4];
    float* out = (float*)d_out;

    const long long N = in_sizes[2];
    const long long K = in_sizes[1] / N;
    const long long M = in_sizes[0] / K;

    unsigned char* Aq = (unsigned char*)d_ws;
    unsigned char* Bq = Aq + (size_t)(M * K);

    {
        long long n4 = (M * K) / 4;
        long long blocks = (n4 + 255) / 256;
        quant_fp8_kernel<<<dim3((unsigned)blocks), 256, 0, stream>>>(
            x, (unsigned int*)Aq, s_in, n4);
    }
    {
        long long n4 = (N * K) / 4;
        long long blocks = (n4 + 255) / 256;
        quant_fp8_kernel<<<dim3((unsigned)blocks), 256, 0, stream>>>(
            weight, (unsigned int*)Bq, s_w, n4);
    }
    {
        dim3 grid((unsigned)(N / 256), (unsigned)(M / 128));
        fp8_wmma_gemm_kernel<<<grid, 256, 0, stream>>>(
            Aq, Bq, bias, s_in, s_w, out, (int)M, (int)N, (int)K);
    }
}
